// Attention_63823214018607
// MI455X (gfx1250) — compile-verified, matrix operand order checked by reading, not run on hardware
//
#include <hip/hip_runtime.h>

// ---------------------------------------------------------------------------
// Types
// ---------------------------------------------------------------------------
typedef unsigned short u16;
typedef __attribute__((ext_vector_type(16))) __bf16 v16bf;
typedef __attribute__((ext_vector_type(8)))  float  v8f;
typedef __attribute__((ext_vector_type(8)))  u16    v8u16;
typedef __attribute__((ext_vector_type(4)))  int    v4i;

#define AS1 __attribute__((address_space(1)))
#define AS3 __attribute__((address_space(3)))

#define B_N   2
#define T_N   2048
#define DIM_N 2048
#define H_N   32
#define HKV_N 4
#define D_N   64
#define M_N   (B_N * T_N)   /* 4096 rows of x */

union FragU { v16bf bf; u16 u[16]; };

__device__ __forceinline__ u16 f32_to_bf16(float f) {
  unsigned u = __float_as_uint(f);
  u += 0x7FFFu + ((u >> 16) & 1u);          // round-to-nearest-even
  return (u16)(u >> 16);
}

// A/B fragment loader for NT WMMA (16x16x32 bf16), row-major source, inner dim k.
// Lane l (<16): row row0+l, k-chunks {k0..k0+7, k0+16..k0+23}
// Lane 16+l   : row row0+l, k-chunks {k0+8..k0+15, k0+24..k0+31}
__device__ __forceinline__ v16bf load_frag_nt(const u16* base, int ld, int row0,
                                              int k0, int lane) {
  const int half = (lane >> 4) & 1;
  const int l    = lane & 15;
  const u16* p = base + (size_t)(row0 + l) * (size_t)ld + (size_t)(k0 + 8 * half);
  v8u16 lo = *(const v8u16*)(p);
  v8u16 hi = *(const v8u16*)(p + 16);
  FragU f;
#pragma unroll
  for (int i = 0; i < 8; ++i) { f.u[i] = lo[i]; f.u[8 + i] = hi[i]; }
  return f.bf;
}

__device__ __forceinline__ v8f wmma_bf16(v16bf a, v16bf b, v8f c) {
  return __builtin_amdgcn_wmma_f32_16x16x32_bf16(false, a, false, b,
                                                 (short)0, c, false, false);
}

// CDNA5 async copy: global -> LDS, 16 bytes, tracked by ASYNCcnt.
// Builtin signature (probe-discovered): (AS1 v4i*, AS3 v4i*, imm offset, imm cpol)
__device__ __forceinline__ void async_copy_b128(const u16* gsrc, u16* ldst) {
#if __has_builtin(__builtin_amdgcn_global_load_async_to_lds_b128)
  __builtin_amdgcn_global_load_async_to_lds_b128(
      (AS1 v4i*)gsrc, (AS3 v4i*)ldst, 0, 0);
#else
  *(v8u16*)ldst = *(const v8u16*)gsrc;       // synchronous fallback
#endif
}

__device__ __forceinline__ void wait_async_le4(void) {
#if __has_builtin(__builtin_amdgcn_s_wait_asynccnt)
  __builtin_amdgcn_s_wait_asynccnt(4);
#else
  asm volatile("s_wait_asynccnt 4" ::: "memory");
#endif
}

__device__ __forceinline__ void wait_async_le0(void) {
#if __has_builtin(__builtin_amdgcn_s_wait_asynccnt)
  __builtin_amdgcn_s_wait_asynccnt(0);
#else
  asm volatile("s_wait_asynccnt 0" ::: "memory");
#endif
}

// ---------------------------------------------------------------------------
// Elementwise prep kernels
// ---------------------------------------------------------------------------
__global__ void cvt_bf16(const float* __restrict__ X, u16* __restrict__ Y, size_t n) {
  size_t i = (size_t)blockIdx.x * blockDim.x + threadIdx.x;
  if (i < n) Y[i] = f32_to_bf16(X[i]);
}

// W[k][n] (KxN, row-major f32) -> Wt[n][k] (NxK, row-major bf16)
__global__ void transpose_to_bf16(const float* __restrict__ W, u16* __restrict__ Wt,
                                  int K, int N) {
  size_t i = (size_t)blockIdx.x * blockDim.x + threadIdx.x;
  if (i >= (size_t)K * N) return;
  int k = (int)(i % K);
  int n = (int)(i / K);
  Wt[i] = f32_to_bf16(W[(size_t)k * N + n]);
}

// ---------------------------------------------------------------------------
// NT WMMA GEMM: C[M,N] (f32) = A[M,K] (bf16 rows) . Bt[N,K] (bf16 rows)
// 256 threads = 8 waves (2M x 4N), block tile 128x128, wave tile 64x32.
// Double-buffered LDS tiles (A:128x32, B:128x32 bf16) filled with
// GLOBAL_LOAD_ASYNC_TO_LDS_B128 (ASYNCcnt), compute reads via ds_load_b128.
// ---------------------------------------------------------------------------
__global__ __launch_bounds__(256)
void gemm_nt_wmma(const u16* __restrict__ A, const u16* __restrict__ Bt,
                  float* __restrict__ C, int M, int N, int K) {
  __shared__ __attribute__((aligned(16))) u16 sA[2][128 * 32];
  __shared__ __attribute__((aligned(16))) u16 sB[2][128 * 32];

  const int lane = threadIdx.x & 31;
  const int wave = threadIdx.x >> 5;
  const int wm = wave >> 2;                 // 0..1
  const int wn = wave & 3;                  // 0..3
  const int mblk = blockIdx.y * 128;
  const int nblk = blockIdx.x * 128;

  // Cooperative copy mapping: 256 threads x 32B = one 8KB tile (2 b128 each).
  const int crow = threadIdx.x & 127;       // tile row 0..127
  const int chof = (threadIdx.x >> 7) << 4; // element offset 0 or 16

  const u16* gA = A  + (size_t)(mblk + crow) * K + chof;
  const u16* gB = Bt + (size_t)(nblk + crow) * K + chof;

  auto issue_copy = [&](int buf, int k0) {
    u16* la = &sA[buf][crow * 32 + chof];
    u16* lb = &sB[buf][crow * 32 + chof];
    async_copy_b128(gA + k0,     la);
    async_copy_b128(gA + k0 + 8, la + 8);
    async_copy_b128(gB + k0,     lb);
    async_copy_b128(gB + k0 + 8, lb + 8);
  };

  const v8f zero = {0.f, 0.f, 0.f, 0.f, 0.f, 0.f, 0.f, 0.f};
  v8f acc[4][2];
#pragma unroll
  for (int mi = 0; mi < 4; ++mi)
#pragma unroll
    for (int ni = 0; ni < 2; ++ni) acc[mi][ni] = zero;

  issue_copy(0, 0);
  int buf = 0;
  for (int k0 = 0; k0 < K; k0 += 32) {
    if (k0 + 32 < K) {                      // overlap next tile copy with compute
      issue_copy(buf ^ 1, k0 + 32);
      wait_async_le4();                     // current tile's 4 copies complete
    } else {
      wait_async_le0();
    }
    __syncthreads();                        // tile visible to all waves

    const u16* la = &sA[buf][0];
    const u16* lb = &sB[buf][0];
    v16bf a[4], b[2];
#pragma unroll
    for (int mi = 0; mi < 4; ++mi)
      a[mi] = load_frag_nt(la, 32, wm * 64 + 16 * mi, 0, lane);
#pragma unroll
    for (int ni = 0; ni < 2; ++ni)
      b[ni] = load_frag_nt(lb, 32, wn * 32 + 16 * ni, 0, lane);
#pragma unroll
    for (int mi = 0; mi < 4; ++mi)
#pragma unroll
      for (int ni = 0; ni < 2; ++ni)
        acc[mi][ni] = wmma_bf16(a[mi], b[ni], acc[mi][ni]);

    __syncthreads();                        // all reads done before buf reuse
    buf ^= 1;
  }

  const int half = (lane >> 4) & 1;
  const int l    = lane & 15;
  const int m0 = mblk + wm * 64;
  const int n0 = nblk + wn * 32;
#pragma unroll
  for (int mi = 0; mi < 4; ++mi)
#pragma unroll
    for (int ni = 0; ni < 2; ++ni)
#pragma unroll
      for (int r = 0; r < 8; ++r)
        C[(size_t)(m0 + 16 * mi + 8 * half + r) * N + (n0 + 16 * ni + l)] =
            acc[mi][ni][r];
}

// ---------------------------------------------------------------------------
// RoPE + layout packing
// ---------------------------------------------------------------------------
__global__ void rope_pack_q(const float* __restrict__ Qf, u16* __restrict__ Qb) {
  size_t i = (size_t)blockIdx.x * blockDim.x + threadIdx.x;
  if (i >= (size_t)M_N * DIM_N) return;
  const int col   = (int)(i & (DIM_N - 1));
  const size_t row = i >> 11;               // b*T + t
  const int t = (int)(row & (T_N - 1));
  const int b = (int)(row >> 11);
  const int h = col >> 6;
  const int d = col & 63;
  const int j = d & 31;
  const float inv = __expf(-(float)j * (9.210340371976184f / 32.0f)); // 10000^(-j/32)
  float s, c;
  __sincosf((float)t * inv, &s, &c);
  const float x0 = Qf[i];
  const float xo = (d < 32) ? Qf[i + 32] : Qf[i - 32];
  const float v  = (d < 32) ? (x0 * c - xo * s) : (x0 * c + xo * s);
  Qb[(((size_t)b * H_N + h) * T_N + t) * D_N + d] = f32_to_bf16(v);
}

__global__ void rope_pack_kv(const float* __restrict__ Kf, const float* __restrict__ Vf,
                             u16* __restrict__ Kb, u16* __restrict__ Vtb) {
  size_t i = (size_t)blockIdx.x * blockDim.x + threadIdx.x;
  if (i >= (size_t)M_N * 256) return;
  const int col   = (int)(i & 255);
  const size_t row = i >> 8;
  const int t  = (int)(row & (T_N - 1));
  const int b  = (int)(row >> 11);
  const int hk = col >> 6;
  const int d  = col & 63;
  const int j  = d & 31;
  const float inv = __expf(-(float)j * (9.210340371976184f / 32.0f));
  float s, c;
  __sincosf((float)t * inv, &s, &c);
  const float x0 = Kf[i];
  const float xo = (d < 32) ? Kf[i + 32] : Kf[i - 32];
  const float v  = (d < 32) ? (x0 * c - xo * s) : (x0 * c + xo * s);
  Kb[(((size_t)b * HKV_N + hk) * T_N + t) * D_N + d]  = f32_to_bf16(v);
  Vtb[(((size_t)b * HKV_N + hk) * D_N + d) * T_N + t] = f32_to_bf16(Vf[i]); // V^T
}

// ---------------------------------------------------------------------------
// Flash attention (online softmax), 1 wave = 16 query rows, k-blocks of 32.
// S = Q.K^T (NT, inner=D), P staged via LDS (C-layout -> A-layout),
// O += P.V^T (NT, inner=k).  Causal mask in-register.
// ---------------------------------------------------------------------------
__global__ __launch_bounds__(128)
void flash_attn_wmma(const u16* __restrict__ Qb, const u16* __restrict__ Kb,
                     const u16* __restrict__ Vtb, u16* __restrict__ Ob) {
  __shared__ __attribute__((aligned(16))) u16 Pl[4][16 * 32];
  const int lane = threadIdx.x & 31;
  const int wave = threadIdx.x >> 5;
  const int half = (lane >> 4) & 1;
  const int l    = lane & 15;
  const int b    = blockIdx.z;
  const int h    = blockIdx.y;
  const int hkv  = h / (H_N / HKV_N);
  const int q0   = (blockIdx.x * 4 + wave) * 16;

  const u16* Q  = Qb  + (((size_t)b * H_N + h) * T_N + q0) * D_N;
  const u16* Kh = Kb  + ((size_t)b * HKV_N + hkv) * (size_t)T_N * D_N;
  const u16* Vt = Vtb + ((size_t)b * HKV_N + hkv) * (size_t)D_N * T_N;
  u16* Pw = &Pl[wave][0];

  const v8f zero = {0.f, 0.f, 0.f, 0.f, 0.f, 0.f, 0.f, 0.f};
  v8f o[4];
#pragma unroll
  for (int dt = 0; dt < 4; ++dt) o[dt] = zero;
  float m_i[8], l_i[8];
#pragma unroll
  for (int r = 0; r < 8; ++r) { m_i[r] = -1e30f; l_i[r] = 0.0f; }

  const v16bf qa0 = load_frag_nt(Q, D_N, 0, 0,  lane);
  const v16bf qa1 = load_frag_nt(Q, D_N, 0, 32, lane);

  const float scale = 0.125f;               // 1/sqrt(64)
  const int kend = (((q0 + 16) + 31) >> 5) << 5;

  for (int k0 = 0; k0 < kend; k0 += 32) {
    if (k0 + 32 < kend) {
      __builtin_prefetch(Kh + (size_t)(k0 + 32 + lane) * D_N, 0, 1);
      __builtin_prefetch(Vt + (size_t)lane * T_N + k0 + 32, 0, 1);
    }
    const v16bf kb0 = load_frag_nt(Kh, D_N, k0,      0,  lane);
    const v16bf kb1 = load_frag_nt(Kh, D_N, k0,      32, lane);
    const v16bf kb2 = load_frag_nt(Kh, D_N, k0 + 16, 0,  lane);
    const v16bf kb3 = load_frag_nt(Kh, D_N, k0 + 16, 32, lane);
    v8f s0 = zero, s1 = zero;
    s0 = wmma_bf16(qa0, kb0, s0);
    s0 = wmma_bf16(qa1, kb1, s0);
    s1 = wmma_bf16(qa0, kb2, s1);
    s1 = wmma_bf16(qa1, kb3, s1);

    float alph[8];
#pragma unroll
    for (int r = 0; r < 8; ++r) {
      const int rowg = q0 + 8 * half + r;
      float a0 = (k0 + l      <= rowg) ? s0[r] * scale : -1e30f;
      float a1 = (k0 + 16 + l <= rowg) ? s1[r] * scale : -1e30f;
      float mx = fmaxf(a0, a1);
#pragma unroll
      for (int msk = 8; msk >= 1; msk >>= 1) mx = fmaxf(mx, __shfl_xor(mx, msk, 32));
      const float mnew = fmaxf(m_i[r], mx);
      const float p0 = __expf(a0 - mnew);
      const float p1 = __expf(a1 - mnew);
      float rs = p0 + p1;
#pragma unroll
      for (int msk = 8; msk >= 1; msk >>= 1) rs += __shfl_xor(rs, msk, 32);
      const float alpha = __expf(m_i[r] - mnew);
      m_i[r] = mnew;
      l_i[r] = l_i[r] * alpha + rs;
      alph[r] = alpha;
      Pw[(8 * half + r) * 32 + l]      = f32_to_bf16(p0);
      Pw[(8 * half + r) * 32 + 16 + l] = f32_to_bf16(p1);
    }
#pragma unroll
    for (int dt = 0; dt < 4; ++dt)
#pragma unroll
      for (int r = 0; r < 8; ++r) o[dt][r] *= alph[r];

    asm volatile("s_wait_dscnt 0" ::: "memory");   // LDS stores visible to frag load
    const v16bf pa = load_frag_nt(Pw, 32, 0, 0, lane);

#pragma unroll
    for (int dt = 0; dt < 4; ++dt) {
      const v16bf vb = load_frag_nt(Vt, T_N, dt * 16, k0, lane);
      o[dt] = wmma_bf16(pa, vb, o[dt]);
    }
  }

#pragma unroll
  for (int dt = 0; dt < 4; ++dt)
#pragma unroll
    for (int r = 0; r < 8; ++r) {
      const int t = q0 + 8 * half + r;
      Ob[((size_t)b * T_N + t) * DIM_N + h * D_N + dt * 16 + l] =
          f32_to_bf16(o[dt][r] / l_i[r]);
    }
}

// ---------------------------------------------------------------------------
// Host launcher
// ---------------------------------------------------------------------------
extern "C" void kernel_launch(void* const* d_in, const int* in_sizes, int n_in,
                              void* d_out, int out_size, void* d_ws, size_t ws_size,
                              hipStream_t stream) {
  (void)in_sizes; (void)n_in; (void)out_size; (void)ws_size;
  const float* x  = (const float*)d_in[0];
  const float* wq = (const float*)d_in[1];
  const float* wk = (const float*)d_in[2];
  const float* wv = (const float*)d_in[3];
  const float* wo = (const float*)d_in[4];
  float* out = (float*)d_out;

  char* base = (char*)d_ws;
  size_t off = 0;
  auto alloc = [&](size_t bytes) -> void* {
    void* p = base + off;
    off = (off + bytes + 255) & ~(size_t)255;
    return p;
  };
  u16*   xb  = (u16*)  alloc((size_t)M_N * DIM_N * 2);
  u16*   wqT = (u16*)  alloc((size_t)DIM_N * DIM_N * 2);
  u16*   wkT = (u16*)  alloc((size_t)256 * DIM_N * 2);
  u16*   wvT = (u16*)  alloc((size_t)256 * DIM_N * 2);
  u16*   woT = (u16*)  alloc((size_t)DIM_N * DIM_N * 2);
  float* qf  = (float*)alloc((size_t)M_N * DIM_N * 4);
  float* kf  = (float*)alloc((size_t)M_N * 256 * 4);
  float* vf  = (float*)alloc((size_t)M_N * 256 * 4);
  u16*   qb  = (u16*)  alloc((size_t)M_N * DIM_N * 2);
  u16*   kb  = (u16*)  alloc((size_t)M_N * 256 * 2);
  u16*   vtb = (u16*)  alloc((size_t)M_N * 256 * 2);
  u16*   ab  = (u16*)  alloc((size_t)M_N * DIM_N * 2);

  const int TB = 256;
  cvt_bf16<<<(M_N * DIM_N) / TB, TB, 0, stream>>>(x, xb, (size_t)M_N * DIM_N);
  transpose_to_bf16<<<(DIM_N * DIM_N) / TB, TB, 0, stream>>>(wq, wqT, DIM_N, DIM_N);
  transpose_to_bf16<<<(DIM_N * 256)   / TB, TB, 0, stream>>>(wk, wkT, DIM_N, 256);
  transpose_to_bf16<<<(DIM_N * 256)   / TB, TB, 0, stream>>>(wv, wvT, DIM_N, 256);
  transpose_to_bf16<<<(DIM_N * DIM_N) / TB, TB, 0, stream>>>(wo, woT, DIM_N, DIM_N);

  gemm_nt_wmma<<<dim3(DIM_N / 128, M_N / 128), 256, 0, stream>>>(xb, wqT, qf, M_N, DIM_N, DIM_N);
  gemm_nt_wmma<<<dim3(256   / 128, M_N / 128), 256, 0, stream>>>(xb, wkT, kf, M_N, 256,   DIM_N);
  gemm_nt_wmma<<<dim3(256   / 128, M_N / 128), 256, 0, stream>>>(xb, wvT, vf, M_N, 256,   DIM_N);

  rope_pack_q <<<(M_N * DIM_N) / TB, TB, 0, stream>>>(qf, qb);
  rope_pack_kv<<<(M_N * 256)   / TB, TB, 0, stream>>>(kf, vf, kb, vtb);

  flash_attn_wmma<<<dim3(T_N / 64, H_N, B_N), 128, 0, stream>>>(qb, kb, vtb, ab);

  gemm_nt_wmma<<<dim3(DIM_N / 128, M_N / 128), 256, 0, stream>>>(ab, woT, out, M_N, DIM_N, DIM_N);
}
